// GAT_Encoder_7627861917897
// MI455X (gfx1250) — matrix operand order repaired and checked by reading, compiled, not run on hardware
//
#include <hip/hip_runtime.h>
#include <hip/hip_bf16.h>

#define Nn   10000
#define Ee   160000
#define ENe  (Ee + Nn)
#define INF_ 256
#define HIDf 128
#define EMBf 256
#define H0f  512   // HEADS*HID
#define HEADSf 4

typedef unsigned short u16;
typedef __attribute__((ext_vector_type(16))) __bf16 v16bf;
typedef __attribute__((ext_vector_type(8)))  float  v8f;

union AFrag { uint4 q[2]; v16bf v; };

// ---- float <-> ordered uint (for atomic float max) ----
__device__ __forceinline__ unsigned f2ord(float f) {
    unsigned u = __float_as_uint(f);
    return (u & 0x80000000u) ? ~u : (u | 0x80000000u);
}
__device__ __forceinline__ float ord2f(unsigned u) {
    return (u & 0x80000000u) ? __uint_as_float(u & 0x7fffffffu) : __uint_as_float(~u);
}
__device__ __forceinline__ float lrelu(float v) { return v >= 0.f ? v : 0.2f * v; }
__device__ __forceinline__ u16 f2bf(float f) {
    unsigned u = __float_as_uint(f);
    return (u16)((u + (0x7fffu + ((u >> 16) & 1u))) >> 16);
}

// ---------------- init ----------------
__global__ void zero_kernel(uint4* p, long n4) {
    long i = (long)blockIdx.x * blockDim.x + threadIdx.x;
    if (i < n4) { uint4 z; z.x = z.y = z.z = z.w = 0u; p[i] = z; }
}

// ---------------- self loops ----------------
__global__ void selfloop_accum(const float* w, const int* dst, float* sum_w, float* cnt) {
    int i = blockIdx.x * blockDim.x + threadIdx.x;
    if (i < Ee) {
        int d = dst[i];
        atomicAdd(&sum_w[d], w[i]);
        atomicAdd(&cnt[d], 1.0f);
    }
}
__global__ void build_ea(const float* w, const float* sum_w, const float* cnt, float* ea) {
    int i = blockIdx.x * blockDim.x + threadIdx.x;
    if (i < ENe) {
        if (i < Ee) ea[i] = w[i];
        else {
            int n = i - Ee;
            float c = cnt[n];
            ea[i] = c > 0.f ? sum_w[n] / fmaxf(c, 1.0f) : 0.0f;
        }
    }
}

// ---------------- f32 -> bf16 row-major (A matrices) ----------------
__global__ void f2bf_kernel(const float* in, u16* out, int n) {
    int i = blockIdx.x * blockDim.x + threadIdx.x;
    if (i < n) out[i] = f2bf(in[i]);
}

// ---------------- pack f32 weight B[K,Nout] -> bf16 WMMA fragment-major ----------------
// Block (tn, ks) holds 512 bf16: [0..255]  = lanes' elems 0..7  (K = ks*32 + 8h + 0..7)
//                                [256..511] = lanes' elems 8..15 (K = ks*32 + 16 + 8h + 0..7)
// lane L: column n = tn*16 + (L&15), h = L>>4.
__global__ void packB_kernel(const float* __restrict__ B, u16* __restrict__ P, int K, int Nout) {
    int idx = blockIdx.x * blockDim.x + threadIdx.x;
    if (idx >= K * Nout) return;
    int t    = idx & 7;
    int lane = (idx >> 3) & 31;
    int half = (idx >> 8) & 1;
    int blk  = idx >> 9;                      // tn * (K/32) + ks
    int Ks = K >> 5;
    int tn = blk / Ks, ks = blk % Ks;
    int i = half * 8 + t;
    int k = ks * 32 + 8 * (lane >> 4) + 16 * half + t;
    int n = tn * 16 + (lane & 15);
    P[idx] = f2bf(B[(size_t)k * Nout + n]);
}

// ---------------- WMMA bf16 GEMM: C[M,Nout] = A[M,K] @ B[K,Nout] + bias ----------------
// One wave per 16x64 C strip (4 N-tiles); A fragment reused across 4 WMMAs per k-step.
__global__ void gemm_bf16_kernel(const u16* __restrict__ A, const u16* __restrict__ P,
                                 const float* __restrict__ bias, float* __restrict__ C,
                                 int M, int K, int Nout) {
    int wid  = (blockIdx.x * blockDim.x + threadIdx.x) >> 5;
    int lane = threadIdx.x & 31;
    int groupsN = Nout >> 6;                  // strips of 4 tiles
    int numW = (M >> 4) * groupsN;
    if (wid >= numW) return;                  // wave-uniform guard (EXEC all-ones inside)
    int tm = wid / groupsN, tg = wid % groupsN;
    int h   = lane >> 4;
    int l16 = lane & 15;
    int Ks  = K >> 5;
    const u16* abase = A + (size_t)(tm * 16 + l16) * K + 8 * h;
    const u16* pbase = P + (size_t)(tg * 4) * Ks * 512 + (size_t)lane * 8;
    const size_t tstr = (size_t)Ks * 512;     // packed elements per n-tile column

    v8f acc0 = 0.0f, acc1 = 0.0f, acc2 = 0.0f, acc3 = 0.0f;
#pragma unroll 2
    for (int ks = 0; ks < Ks; ++ks) {
        AFrag af;
        af.q[0] = *(const uint4*)(abase + ks * 32);
        af.q[1] = *(const uint4*)(abase + ks * 32 + 16);
        const u16* pk = pbase + (size_t)ks * 512;
        AFrag b0, b1, b2, b3;
        b0.q[0] = *(const uint4*)(pk);
        b0.q[1] = *(const uint4*)(pk + 256);
        b1.q[0] = *(const uint4*)(pk + tstr);
        b1.q[1] = *(const uint4*)(pk + tstr + 256);
        b2.q[0] = *(const uint4*)(pk + 2 * tstr);
        b2.q[1] = *(const uint4*)(pk + 2 * tstr + 256);
        b3.q[0] = *(const uint4*)(pk + 3 * tstr);
        b3.q[1] = *(const uint4*)(pk + 3 * tstr + 256);
        acc0 = __builtin_amdgcn_wmma_f32_16x16x32_bf16(false, af.v, false, b0.v, (short)0, acc0, false, false);
        acc1 = __builtin_amdgcn_wmma_f32_16x16x32_bf16(false, af.v, false, b1.v, (short)0, acc1, false, false);
        acc2 = __builtin_amdgcn_wmma_f32_16x16x32_bf16(false, af.v, false, b2.v, (short)0, acc2, false, false);
        acc3 = __builtin_amdgcn_wmma_f32_16x16x32_bf16(false, af.v, false, b3.v, (short)0, acc3, false, false);
    }
#define STORE_TILE(ACC, J)                                                        \
    {                                                                             \
        int ncol = (tg * 4 + (J)) * 16 + l16;                                     \
        float bv = bias[ncol];                                                    \
        _Pragma("unroll")                                                         \
        for (int v = 0; v < 8; ++v)                                               \
            C[(size_t)(tm * 16 + v + 8 * h) * Nout + ncol] = ACC[v] + bv;         \
    }
    STORE_TILE(acc0, 0)
    STORE_TILE(acc1, 1)
    STORE_TILE(acc2, 2)
    STORE_TILE(acc3, 3)
#undef STORE_TILE
}

// ---------------- Layer 0 edges: logits + segment max (wave per edge, 512 ch, 4 heads) ----------------
__global__ void edge0_kernel(const int* src, const int* dst, const float* ea,
                             const float* __restrict__ xl, const float* __restrict__ xr,
                             const float* __restrict__ We, const float* __restrict__ att,
                             float* logits, unsigned* amax) {
    int wid = (int)(((long)blockIdx.x * blockDim.x + threadIdx.x) >> 5);
    if (wid >= ENe) return;
    int lane = threadIdx.x & 31;
    int s = wid < Ee ? src[wid] : (wid - Ee);
    int d = wid < Ee ? dst[wid] : (wid - Ee);
    float w = ea[wid];
    int c0 = lane * 16;
    const float4* xlp = (const float4*)(xl + (size_t)s * H0f + c0);
    const float4* xrp = (const float4*)(xr + (size_t)d * H0f + c0);
    const float4* wep = (const float4*)(We + c0);
    const float4* atp = (const float4*)(att + c0);
    float part = 0.f;
#pragma unroll
    for (int j = 0; j < 4; ++j) {
        float4 a = xlp[j], b = xrp[j], we = wep[j], at = atp[j];
        float m;
        m = lrelu(a.x + b.x + w * we.x); part += m * at.x;
        m = lrelu(a.y + b.y + w * we.y); part += m * at.y;
        m = lrelu(a.z + b.z + w * we.z); part += m * at.z;
        m = lrelu(a.w + b.w + w * we.w); part += m * at.w;
    }
    part += __shfl_xor(part, 4, 32);
    part += __shfl_xor(part, 2, 32);
    part += __shfl_xor(part, 1, 32);
    int head = lane >> 3;
    if ((lane & 7) == 0) {
        logits[(size_t)wid * HEADSf + head] = part;
        atomicMax(&amax[d * HEADSf + head], f2ord(part));
    }
}

__global__ void ex0_kernel(const int* dst, float* lg, const unsigned* amax, float* denom) {
    int i = blockIdx.x * blockDim.x + threadIdx.x;
    if (i >= ENe * HEADSf) return;
    int e = i >> 2, h = i & 3;
    int d = e < Ee ? dst[e] : (e - Ee);
    float ex = __expf(lg[i] - ord2f(amax[d * HEADSf + h]));
    lg[i] = ex;
    atomicAdd(&denom[d * HEADSf + h], ex);
}

__global__ void agg0_kernel(const int* src, const int* dst, const float* ex, const float* denom,
                            const float* __restrict__ xl, float* out) {
    int wid = (int)(((long)blockIdx.x * blockDim.x + threadIdx.x) >> 5);
    if (wid >= ENe) return;
    int lane = threadIdx.x & 31;
    int s = wid < Ee ? src[wid] : (wid - Ee);
    int d = wid < Ee ? dst[wid] : (wid - Ee);
    int c0 = lane * 16, head = lane >> 3;
    float alpha = ex[(size_t)wid * HEADSf + head] / (denom[d * HEADSf + head] + 1e-16f);
    const float* xs = xl + (size_t)s * H0f + c0;
    float* od = out + (size_t)d * H0f + c0;
#pragma unroll
    for (int t = 0; t < 16; ++t) atomicAdd(&od[t], alpha * xs[t]);
}

// bias0 + PReLU -> bf16 feature for layer-1 GEMMs
__global__ void post0_kernel(const float* out0, const float* bias0, const float* prelu0, u16* h0bf) {
    int i = blockIdx.x * blockDim.x + threadIdx.x;
    if (i >= Nn * H0f) return;
    int c = i & (H0f - 1);
    float v = out0[i] + bias0[c];
    v = v >= 0.f ? v : prelu0[c] * v;
    h0bf[i] = f2bf(v);
}

// ---------------- Layer 1 edges (1 head, 256 ch) ----------------
__global__ void edge1_kernel(const int* src, const int* dst, const float* ea,
                             const float* __restrict__ xl, const float* __restrict__ xr,
                             const float* __restrict__ We, const float* __restrict__ att,
                             float* logits, unsigned* amax) {
    int wid = (int)(((long)blockIdx.x * blockDim.x + threadIdx.x) >> 5);
    if (wid >= ENe) return;
    int lane = threadIdx.x & 31;
    int s = wid < Ee ? src[wid] : (wid - Ee);
    int d = wid < Ee ? dst[wid] : (wid - Ee);
    float w = ea[wid];
    int c0 = lane * 8;
    const float4* xlp = (const float4*)(xl + (size_t)s * EMBf + c0);
    const float4* xrp = (const float4*)(xr + (size_t)d * EMBf + c0);
    const float4* wep = (const float4*)(We + c0);
    const float4* atp = (const float4*)(att + c0);
    float part = 0.f;
#pragma unroll
    for (int j = 0; j < 2; ++j) {
        float4 a = xlp[j], b = xrp[j], we = wep[j], at = atp[j];
        float m;
        m = lrelu(a.x + b.x + w * we.x); part += m * at.x;
        m = lrelu(a.y + b.y + w * we.y); part += m * at.y;
        m = lrelu(a.z + b.z + w * we.z); part += m * at.z;
        m = lrelu(a.w + b.w + w * we.w); part += m * at.w;
    }
    part += __shfl_xor(part, 16, 32);
    part += __shfl_xor(part, 8, 32);
    part += __shfl_xor(part, 4, 32);
    part += __shfl_xor(part, 2, 32);
    part += __shfl_xor(part, 1, 32);
    if (lane == 0) {
        logits[wid] = part;
        atomicMax(&amax[d], f2ord(part));
    }
}

__global__ void ex1_kernel(const int* dst, float* lg, const unsigned* amax, float* denom) {
    int i = blockIdx.x * blockDim.x + threadIdx.x;
    if (i >= ENe) return;
    int d = i < Ee ? dst[i] : (i - Ee);
    float ex = __expf(lg[i] - ord2f(amax[d]));
    lg[i] = ex;
    atomicAdd(&denom[d], ex);
}

__global__ void agg1_kernel(const int* src, const int* dst, const float* ex, const float* denom,
                            const float* __restrict__ xl, float* out) {
    int wid = (int)(((long)blockIdx.x * blockDim.x + threadIdx.x) >> 5);
    if (wid >= ENe) return;
    int lane = threadIdx.x & 31;
    int s = wid < Ee ? src[wid] : (wid - Ee);
    int d = wid < Ee ? dst[wid] : (wid - Ee);
    int c0 = lane * 8;
    float alpha = ex[wid] / (denom[d] + 1e-16f);
    const float* xs = xl + (size_t)s * EMBf + c0;
    float* od = out + (size_t)d * EMBf + c0;
#pragma unroll
    for (int t = 0; t < 8; ++t) atomicAdd(&od[t], alpha * xs[t]);
}

// out = prelu(out1 + bias1 + skip, prelu1)
__global__ void final_kernel(const float* out1, const float* skip, const float* bias1,
                             const float* prelu1, float* out) {
    int i = blockIdx.x * blockDim.x + threadIdx.x;
    if (i >= Nn * EMBf) return;
    int c = i & (EMBf - 1);
    float v = out1[i] + bias1[c] + skip[i];
    out[i] = v >= 0.f ? v : prelu1[c] * v;
}

extern "C" void kernel_launch(void* const* d_in, const int* in_sizes, int n_in,
                              void* d_out, int out_size, void* d_ws, size_t ws_size,
                              hipStream_t stream) {
    const float* x      = (const float*)d_in[0];
    const float* ew     = (const float*)d_in[1];
    const float* Wl0    = (const float*)d_in[2];
    const float* bl0    = (const float*)d_in[3];
    const float* Wr0    = (const float*)d_in[4];
    const float* br0    = (const float*)d_in[5];
    const float* We0    = (const float*)d_in[6];
    const float* att0   = (const float*)d_in[7];
    const float* bias0  = (const float*)d_in[8];
    const float* Wl1    = (const float*)d_in[9];
    const float* bl1    = (const float*)d_in[10];
    const float* Wr1    = (const float*)d_in[11];
    const float* br1    = (const float*)d_in[12];
    const float* We1    = (const float*)d_in[13];
    const float* att1   = (const float*)d_in[14];
    const float* bias1  = (const float*)d_in[15];
    const float* skipW  = (const float*)d_in[16];
    const float* skipb  = (const float*)d_in[17];
    const float* prelu0 = (const float*)d_in[18];
    const float* prelu1 = (const float*)d_in[19];
    const int*   ei     = (const int*)d_in[20];
    const int* srcA = ei;
    const int* dstA = ei + Ee;
    float* outp = (float*)d_out;

    // ---- bump allocator over d_ws ----
    size_t off = 0;
    auto alloc = [&](size_t bytes) -> void* {
        size_t cur = (off + 255) & ~(size_t)255;
        off = cur + bytes;
        return (void*)((char*)d_ws + cur);
    };

    // contiguous zero-initialized accumulator region
    size_t zfloats = (size_t)Nn * (1 + 1 + 4 + 4 + H0f + 1 + 1 + EMBf);   // 780*N
    float*    zbase  = (float*)alloc(zfloats * 4);
    float*    sum_w  = zbase;
    float*    cntb   = sum_w + Nn;
    unsigned* amax0  = (unsigned*)(cntb + Nn);
    float*    denom0 = (float*)(amax0 + 4 * Nn);
    float*    out0   = denom0 + 4 * Nn;
    unsigned* amax1  = (unsigned*)(out0 + (size_t)Nn * H0f);
    float*    denom1 = (float*)(amax1 + Nn);
    float*    out1   = denom1 + Nn;

    float* ea   = (float*)alloc((size_t)ENe * 4);
    float* lg0  = (float*)alloc((size_t)ENe * HEADSf * 4);
    float* lg1  = (float*)alloc((size_t)ENe * 4);
    float* xl0  = (float*)alloc((size_t)Nn * H0f * 4);
    float* xr0  = (float*)alloc((size_t)Nn * H0f * 4);
    float* xl1  = (float*)alloc((size_t)Nn * EMBf * 4);
    float* xr1  = (float*)alloc((size_t)Nn * EMBf * 4);
    float* skb  = (float*)alloc((size_t)Nn * EMBf * 4);
    u16* x_bf    = (u16*)alloc((size_t)Nn * INF_ * 2);
    u16* h0_bf   = (u16*)alloc((size_t)Nn * H0f * 2);
    u16* Wl0_pk  = (u16*)alloc((size_t)INF_ * H0f * 2);
    u16* Wr0_pk  = (u16*)alloc((size_t)INF_ * H0f * 2);
    u16* Wl1_pk  = (u16*)alloc((size_t)H0f * EMBf * 2);
    u16* Wr1_pk  = (u16*)alloc((size_t)H0f * EMBf * 2);
    u16* skW_pk  = (u16*)alloc((size_t)INF_ * EMBf * 2);

    const int TB = 256;
    // 1. zero accumulators
    long n4 = (long)(zfloats / 4);
    zero_kernel<<<(int)((n4 + TB - 1) / TB), TB, 0, stream>>>((uint4*)zbase, n4);
    // 2-3. self loops + ea
    selfloop_accum<<<(Ee + TB - 1) / TB, TB, 0, stream>>>(ew, dstA, sum_w, cntb);
    build_ea<<<(ENe + TB - 1) / TB, TB, 0, stream>>>(ew, sum_w, cntb, ea);
    // 4. bf16 packs: A-side row-major convert + B-side fragment-major packs
    f2bf_kernel<<<((Nn * INF_) + TB - 1) / TB, TB, 0, stream>>>(x, x_bf, Nn * INF_);
    packB_kernel<<<((INF_ * H0f) + TB - 1) / TB, TB, 0, stream>>>(Wl0, Wl0_pk, INF_, H0f);
    packB_kernel<<<((INF_ * H0f) + TB - 1) / TB, TB, 0, stream>>>(Wr0, Wr0_pk, INF_, H0f);
    packB_kernel<<<((H0f * EMBf) + TB - 1) / TB, TB, 0, stream>>>(Wl1, Wl1_pk, H0f, EMBf);
    packB_kernel<<<((H0f * EMBf) + TB - 1) / TB, TB, 0, stream>>>(Wr1, Wr1_pk, H0f, EMBf);
    packB_kernel<<<((INF_ * EMBf) + TB - 1) / TB, TB, 0, stream>>>(skipW, skW_pk, INF_, EMBf);
    // 5. layer-0 projections (WMMA, 16x64 strip per wave)
    int waves0 = (Nn / 16) * (H0f / 64);            // 5000 waves
    gemm_bf16_kernel<<<(waves0 * 32 + TB - 1) / TB, TB, 0, stream>>>(x_bf, Wl0_pk, bl0, xl0, Nn, INF_, H0f);
    gemm_bf16_kernel<<<(waves0 * 32 + TB - 1) / TB, TB, 0, stream>>>(x_bf, Wr0_pk, br0, xr0, Nn, INF_, H0f);
    // 6-8. layer-0 edge phase
    int eblocks = (ENe * 32 + TB - 1) / TB;
    edge0_kernel<<<eblocks, TB, 0, stream>>>(srcA, dstA, ea, xl0, xr0, We0, att0, lg0, amax0);
    ex0_kernel<<<(ENe * HEADSf + TB - 1) / TB, TB, 0, stream>>>(dstA, lg0, amax0, denom0);
    agg0_kernel<<<eblocks, TB, 0, stream>>>(srcA, dstA, lg0, denom0, xl0, out0);
    // 9. bias + PReLU -> bf16
    post0_kernel<<<((Nn * H0f) + TB - 1) / TB, TB, 0, stream>>>(out0, bias0, prelu0, h0_bf);
    // 10. layer-1 projections + skip (WMMA)
    int waves1 = (Nn / 16) * (EMBf / 64);           // 2500 waves
    gemm_bf16_kernel<<<(waves1 * 32 + TB - 1) / TB, TB, 0, stream>>>(h0_bf, Wl1_pk, bl1, xl1, Nn, H0f, EMBf);
    gemm_bf16_kernel<<<(waves1 * 32 + TB - 1) / TB, TB, 0, stream>>>(h0_bf, Wr1_pk, br1, xr1, Nn, H0f, EMBf);
    gemm_bf16_kernel<<<(waves1 * 32 + TB - 1) / TB, TB, 0, stream>>>(x_bf, skW_pk, skipb, skb, Nn, INF_, EMBf);
    // 11. layer-1 edge phase
    edge1_kernel<<<eblocks, TB, 0, stream>>>(srcA, dstA, ea, xl1, xr1, We1, att1, lg1, amax1);
    ex1_kernel<<<(ENe + TB - 1) / TB, TB, 0, stream>>>(dstA, lg1, amax1, denom1);
    agg1_kernel<<<eblocks, TB, 0, stream>>>(srcA, dstA, lg1, denom1, xl1, out1);
    // 12. skip + bias + PReLU
    final_kernel<<<((Nn * EMBf) + TB - 1) / TB, TB, 0, stream>>>(out1, skb, bias1, prelu1, outp);
}
